// LatticeGaussian_19018115186783
// MI455X (gfx1250) — matrix operand order repaired and checked by reading, not compiled, not run on hardware
//
#include <hip/hip_runtime.h>
#include <hip/hip_bf16.h>

typedef __attribute__((ext_vector_type(16))) _Float16 v16h;
typedef __attribute__((ext_vector_type(8)))  float    v8f;
typedef __attribute__((ext_vector_type(4)))  float    v4f;

#define NN 8192
#define CC 16
#define DD 5

// sqrt(log2(e)) and log2(e): fold exp(x) -> exp2 domain at prep time so the
// inner loop needs exactly one v_exp_f32 per W element, no extra multiply.
#define SQRT_LOG2E 1.2011224087864498f
#define LOG2E      1.4426950408889634f

// Prep: refp[j][0..4] = ref[j][:]*sqrt(log2e); refp[j][5] = -0.5*||ref_j||^2*log2e
//       (8 floats per row -> two aligned B128 loads later)
//       Ut[c][j] = (f16) U[j][c]   (so WMMA B-operand loads are contiguous)
__global__ __launch_bounds__(256)
void lattice_prep_kernel(const float* __restrict__ U,
                         const float* __restrict__ ref,
                         float* __restrict__ refp,
                         _Float16* __restrict__ Ut) {
    int j = blockIdx.x * blockDim.x + threadIdx.x;
    if (j >= NN) return;
    float r[DD];
    float sq = 0.0f;
#pragma unroll
    for (int d = 0; d < DD; ++d) { r[d] = ref[j * DD + d]; sq += r[d] * r[d]; }
    float* o = refp + j * 8;
#pragma unroll
    for (int d = 0; d < DD; ++d) o[d] = r[d] * SQRT_LOG2E;
    o[5] = -0.5f * sq * LOG2E;
    o[6] = 0.0f;
    o[7] = 0.0f;
#pragma unroll
    for (int c = 0; c < CC; ++c) Ut[c * NN + j] = (_Float16)U[j * CC + c];
}

// Main: one block per 16-row tile of the output. 8 waves split the K=8192
// dimension into 32-wide tiles. Each wave builds the f16 A-tile (W values)
// in-register in contiguous k order (k = 16*laneHi + h, same as B) and issues
// v_wmma_f32_16x16x32_f16 against the preloaded U^T B-tile.
__global__ __launch_bounds__(256)
void lattice_main_kernel(const float* __restrict__ U,
                         const float* __restrict__ refp,
                         const _Float16* __restrict__ Ut,
                         float* __restrict__ out) {
    __shared__ float sm[8][256];

    const int tid    = threadIdx.x;
    const int wave   = tid >> 5;
    const int lane   = tid & 31;
    const int laneHi = (lane >> 4) & 1;   // 0: lanes 0-15, 1: lanes 16-31
    const int lane15 = lane & 15;
    const int rbase  = blockIdx.x * 16;
    const int r      = rbase + lane15;    // A-matrix row for this lane

    // This lane's ref row (scaled) + its -0.5*sq*log2e term
    const v4f rp0 = *(const v4f*)(refp + r * 8);
    const v4f rp1 = *(const v4f*)(refp + r * 8 + 4);
    const float a_r = rp1.y;              // element [5]

    v8f acc = {};

    for (int it = 0; it < 32; ++it) {
        const int jbase = (it * 8 + wave) * 32;

        // B operand: U^T[c = lane15][jbase + laneHi*16 + h], h = 0..15
        // (B 32x16 f16, contiguous k order as in A: lanes 0-15 hold K=0..15, lanes 16-31 K=16..31)
        const _Float16* bp = Ut + lane15 * NN + jbase + laneHi * 16;
        v16h b = *(const v16h*)bp;        // 32 contiguous bytes

        // A operand: W[r][jbase + k] in the 16x32 f16 A layout:
        // half h -> k = laneHi*16 + h
        v16h a = {};
#pragma unroll
        for (int h = 0; h < 16; ++h) {
            const int k = laneHi * 16 + h;
            const float* jp = refp + (size_t)(jbase + k) * 8;
            const v4f q0 = *(const v4f*)jp;
            const v4f q1 = *(const v4f*)(jp + 4);
            float t = a_r + q1.y
                    + rp0.x * q0.x + rp0.y * q0.y + rp0.z * q0.z
                    + rp0.w * q0.w + rp1.x * q1.x;
            float w = __builtin_amdgcn_exp2f(t);   // single v_exp_f32
            a[h] = (_Float16)w;
        }

        // D = A(16x32 f16) x B(32x16 f16) + C(16x16 f32)
        acc = __builtin_amdgcn_wmma_f32_16x16x32_f16(
            /*neg_a=*/false, a, /*neg_b=*/false, b,
            /*c_mod=*/(short)0, acc, /*reuse_a=*/false, /*reuse_b=*/false);
    }

    // Cross-wave reduction of the 8 partial C tiles through LDS.
#pragma unroll
    for (int v = 0; v < 8; ++v) sm[wave][lane * 8 + v] = acc[v];
    __syncthreads();

    // One thread per C element: (L, v) -> (m = v + (L>=16 ? 8:0), n = L&15)
    const int L = tid & 31;
    const int v = tid >> 5;
    float s = 0.0f;
#pragma unroll
    for (int w = 0; w < 8; ++w) s += sm[w][L * 8 + v];
    const int m   = v + ((L & 16) ? 8 : 0);
    const int n   = L & 15;
    const int row = rbase + m;
    out[row * CC + n] = s - U[row * CC + n];
}

extern "C" void kernel_launch(void* const* d_in, const int* in_sizes, int n_in,
                              void* d_out, int out_size, void* d_ws, size_t ws_size,
                              hipStream_t stream) {
    const float* U   = (const float*)d_in[0];   // [8192,16] f32
    const float* ref = (const float*)d_in[1];   // [8192,5]  f32
    float* out = (float*)d_out;                 // [8192,16] f32

    // Workspace layout: refp (8192*8 f32 = 256KB) | Ut f16 (16*8192 = 256KB)
    float*    refp = (float*)d_ws;
    _Float16* Ut   = (_Float16*)((char*)d_ws + (size_t)NN * 8 * sizeof(float));

    lattice_prep_kernel<<<NN / 256, 256, 0, stream>>>(U, ref, refp, Ut);
    lattice_main_kernel<<<NN / 16, 256, 0, stream>>>(U, refp, Ut, out);
}
